// RESORT_6743098655210
// MI455X (gfx1250) — matrix operand order repaired and checked by reading, not compiled, not run on hardware
//
#include <hip/hip_runtime.h>

typedef unsigned short ushort_t;
typedef __attribute__((ext_vector_type(16))) __bf16 bf16x16;
typedef __attribute__((ext_vector_type(8)))  float  f32x8;

#define B_SZ   512
#define L_SZ   25
#define D_MSR  256
#define HEADS  8
#define HD     32
#define BL     (B_SZ * L_SZ)      // 12800
#define FUSD   (L_SZ * D_MSR)     // 6400
#define HXY    800                // FUSD / HEADS
#define HXYPAD 896                // HXY padded to multiple of 128

__device__ __forceinline__ float b2f(ushort_t h) {
    return __uint_as_float(((unsigned)h) << 16);
}
__device__ __forceinline__ ushort_t f2b(float f) {
    unsigned u = __float_as_uint(f);
    u += 0x7FFFu + ((u >> 16) & 1u);          // round-to-nearest-even
    return (ushort_t)(u >> 16);
}

// ---------------------------------------------------------------------------
// Convert + transpose weights: W[K x N] f32 row-major -> Wt[Npad x K] bf16,
// zero-filling pad rows so the GEMM needs no bound checks on B.
// ---------------------------------------------------------------------------
__global__ void cvt_tr_kernel(const float* __restrict__ W,
                              ushort_t* __restrict__ Wt,
                              int K, int N, int Npad) {
    int idx = blockIdx.x * blockDim.x + threadIdx.x;
    if (idx >= Npad * K) return;
    int n = idx / K, k = idx - n * K;
    Wt[idx] = (n < N) ? f2b(W[(size_t)k * N + n]) : (ushort_t)0;
}

// ---------------------------------------------------------------------------
// FS gate: g[j] = 2*sigmoid(relu(ctx@W1 + b1) @ W2 + b2)   (25 outputs)
// ---------------------------------------------------------------------------
__global__ void gate_kernel(const float* __restrict__ ctx,
                            const float* __restrict__ W1, const float* __restrict__ b1,
                            const float* __restrict__ W2, const float* __restrict__ b2,
                            float* __restrict__ g) {
    __shared__ float h[64];
    int t = threadIdx.x;                       // 64 threads
    if (t < 64) {
        float s = b1[t];
        for (int c = 0; c < 128; ++c) s += ctx[c] * W1[c * 64 + t];
        h[t] = fmaxf(s, 0.f);
    }
    __syncthreads();
    if (t < L_SZ) {
        float s = b2[t];
        for (int i = 0; i < 64; ++i) s += h[i] * W2[i * L_SZ + t];
        g[t] = 2.f / (1.f + __expf(-s));
    }
}

// ---------------------------------------------------------------------------
// Embedding gather + dense proj, scaled by gates -> bf16 X1/X2 (B*L x 128)
// ---------------------------------------------------------------------------
__global__ void embed_kernel(const int* __restrict__ ids,
                             const float* __restrict__ densef,
                             const float* __restrict__ emb,
                             const float* __restrict__ dW,
                             const float* __restrict__ db,
                             const float* __restrict__ g1,
                             const float* __restrict__ g2,
                             ushort_t* __restrict__ X1,
                             ushort_t* __restrict__ X2) {
    int bl = blockIdx.x;                       // b*25 + f
    int b = bl / L_SZ, f = bl % L_SZ;
    int d = threadIdx.x;                       // 128
    float val;
    if (f < 24) {
        int id = ids[b * 24 + f];
        val = emb[((size_t)f * 10000 + id) * 128 + d];
    } else {
        float s = db[d];
        for (int c = 0; c < 13; ++c) s += densef[b * 13 + c] * dW[c * 128 + d];
        val = s;
    }
    size_t o = (size_t)bl * 128 + d;
    X1[o] = f2b(val * g1[f]);
    X2[o] = f2b(val * g2[f]);
}

// ---------------------------------------------------------------------------
// bf16 WMMA GEMM: C[MxN] f32 = A[MxK] bf16 (row stride lda) @ B[KxN] bf16,
// where B is supplied pre-transposed + padded as Wt[Npad x K] bf16.
// Block tile 128x128, 8 waves (4 in M x 2 in N), each wave 32x64
// -> 8x v_wmma_f32_16x16x32_bf16 per K-step. A staged in LDS; B fragments
// loaded straight from global (16 contiguous K halves per lane = 2x b128).
// Requires M%128==0, K%32==0, Npad%128==0. Only the C store guards n<N.
// ---------------------------------------------------------------------------
#define BM 128
#define BN 128
#define BK 32
#define LPAD 8

union AFrag { ushort_t u[16]; bf16x16 v; };

__global__ __launch_bounds__(256) void gemm_bf16_kernel(
        const ushort_t* __restrict__ A, int lda,
        const ushort_t* __restrict__ Wt,          // [Npad][K]
        float* __restrict__ C, int ldc,
        int M, int N, int K) {
    __shared__ __align__(16) ushort_t As[BM][BK + LPAD];

    int tid  = threadIdx.x;
    int wave = tid >> 5, lane = tid & 31;
    int wr = wave & 3, wc = wave >> 2;            // wave: 4 (M) x 2 (N)
    int m0 = blockIdx.x * BM;
    int n0 = blockIdx.y * BN;
    int lm = lane & 15, lh = lane >> 4;
    int kbA = lh * 16;                            // A: lanes16-31 hold K=16..31
    int kbB = lh * 16;                            // B: lanes16-31 hold K=16..31

    f32x8 acc[2][4] = {};

    // per-lane B row pointers (column n of original B, contiguous in K)
    const ushort_t* bp[4];
#pragma unroll
    for (int ni = 0; ni < 4; ++ni)
        bp[ni] = Wt + (size_t)(n0 + wc * 64 + ni * 16 + lm) * K + kbB;

    // A-tile loader indices: 16 halves per thread
    int lr = tid >> 1, lpart = (tid & 1) * 16;

    for (int k0 = 0; k0 < K; k0 += BK) {
        {
            const ushort_t* src = A + (size_t)(m0 + lr) * lda + k0 + lpart;
            ushort_t* dst = &As[lr][lpart];
            *(uint4*)(dst)     = *(const uint4*)(src);
            *(uint4*)(dst + 8) = *(const uint4*)(src + 8);
        }
        __syncthreads();

        AFrag a[2], bfr[4];
#pragma unroll
        for (int mi = 0; mi < 2; ++mi) {
            const ushort_t* pa = &As[wr * 32 + mi * 16 + lm][0];
            *(uint4*)&a[mi].u[0] = *(const uint4*)(pa + kbA);
            *(uint4*)&a[mi].u[8] = *(const uint4*)(pa + 8 + kbA);
        }
#pragma unroll
        for (int ni = 0; ni < 4; ++ni) {
            *(uint4*)&bfr[ni].u[0] = *(const uint4*)(bp[ni] + k0);
            *(uint4*)&bfr[ni].u[8] = *(const uint4*)(bp[ni] + k0 + 8);
        }
#pragma unroll
        for (int mi = 0; mi < 2; ++mi)
#pragma unroll
            for (int ni = 0; ni < 4; ++ni)
                acc[mi][ni] = __builtin_amdgcn_wmma_f32_16x16x32_bf16(
                    false, a[mi].v, false, bfr[ni].v,
                    (short)0, acc[mi][ni], false, false);
        __syncthreads();
    }

    // store C (f32 16x16 C/D layout: vgpr r -> M = r + 8*(lane>=16))
#pragma unroll
    for (int mi = 0; mi < 2; ++mi)
#pragma unroll
        for (int ni = 0; ni < 4; ++ni) {
            int n = n0 + wc * 64 + ni * 16 + lm;
            if (n < N) {
#pragma unroll
                for (int r = 0; r < 8; ++r) {
                    int m = m0 + wr * 32 + mi * 16 + r + lh * 8;
                    C[(size_t)m * ldc + n] = acc[mi][ni][r];
                }
            }
        }
}

// ---------------------------------------------------------------------------
// xPos rotary, in place on (B*L, 256) f32. One block per (b,l) row.
// ---------------------------------------------------------------------------
__global__ void xpos_kernel(float* __restrict__ X) {
    __shared__ float row[D_MSR];
    int bl = blockIdx.x;
    int l  = bl % L_SZ;
    int d  = threadIdx.x;
    float x = X[(size_t)bl * D_MSR + d];
    row[d] = x;
    __syncthreads();
    int i = d >> 1;                               // pair index 0..127
    float base  = (2.f * i + 0.4f * 256.f) / (1.4f * 256.f);
    float scale = __powf(base, (float)l / 512.f);
    float invf  = __powf(10000.f, -(float)i / 128.f);
    float ang   = (float)l * invf;
    float sn = __sinf(ang) * scale, cs = __cosf(ang) * scale;
    float rot = (d & 1) ? row[d - 1] : -row[d + 1];
    X[(size_t)bl * D_MSR + d] = x * cs + rot * sn;
}

// ---------------------------------------------------------------------------
// Global group norm (stats over whole batch per group of 32 channels)
// ---------------------------------------------------------------------------
__global__ void zero_stats_kernel(float* __restrict__ stats) {
    if (threadIdx.x < 16) stats[threadIdx.x] = 0.f;
}

__global__ void gn_stat_kernel(const float* __restrict__ Kt,
                               float* __restrict__ stats) {
    int bl = blockIdx.x, d = threadIdx.x;         // 256 threads; warp == group
    float v = Kt[(size_t)bl * D_MSR + d];
    float s = v, q = v * v;
    for (int o = 16; o > 0; o >>= 1) {
        s += __shfl_down(s, o, 32);
        q += __shfl_down(q, o, 32);
    }
    if ((d & 31) == 0) {
        int g = d >> 5;
        atomicAdd(&stats[g], s);
        atomicAdd(&stats[8 + g], q);
    }
}

__global__ void gn_apply_kernel(float* __restrict__ Kt,
                                const float* __restrict__ stats,
                                const float* __restrict__ w,
                                const float* __restrict__ b) {
    int bl = blockIdx.x, d = threadIdx.x;
    int g = d >> 5;
    const float Ng = (float)BL * 32.f;            // 409600
    float mean = stats[g] / Ng;
    float var  = stats[8 + g] / Ng - mean * mean;
    float v = Kt[(size_t)bl * D_MSR + d];
    v = (v - mean) * rsqrtf(var + 1e-5f);
    Kt[(size_t)bl * D_MSR + d] = v * w[d] + b[d];
}

// ---------------------------------------------------------------------------
// Retention attention per (b,h): scores = swish(q)@k^T * D ; out = scores@v
// Epilogue: relu(out + res) -> bf16 next-layer activations.
// ---------------------------------------------------------------------------
__global__ __launch_bounds__(32) void attn_kernel(
        const float* __restrict__ Q, const float* __restrict__ Kt,
        const float* __restrict__ V, const float* __restrict__ Res,
        const ushort_t* __restrict__ Xin, int hasRes,
        float gamma, ushort_t* __restrict__ Xout) {
    __shared__ float sq[L_SZ][HD + 1], sk[L_SZ][HD + 1], sv[L_SZ][HD + 1];
    int blk = blockIdx.x;
    int b = blk >> 3, h = blk & 7;
    int d = threadIdx.x;                           // 32 = head dim
    size_t base = (size_t)b * L_SZ * D_MSR + h * HD + d;
    for (int i = 0; i < L_SZ; ++i) {
        float q = Q[base + (size_t)i * D_MSR];
        sq[i][d] = q / (1.f + __expf(-q));         // swish(q)
        sk[i][d] = Kt[base + (size_t)i * D_MSR];
        sv[i][d] = V[base + (size_t)i * D_MSR];
    }
    __syncthreads();
    if (d < L_SZ) {
        int i = d;
        float out[HD];
#pragma unroll
        for (int c = 0; c < HD; ++c) out[c] = 0.f;
        float gpow = 1.f;
        for (int j = i; j >= 0; --j) {             // gamma^(i-j) decay
            float s = 0.f;
#pragma unroll
            for (int c = 0; c < HD; ++c) s += sq[i][c] * sk[j][c];
            s *= gpow;
#pragma unroll
            for (int c = 0; c < HD; ++c) out[c] += s * sv[j][c];
            gpow *= gamma;
        }
        size_t ob = ((size_t)b * L_SZ + i) * D_MSR + h * HD;
        for (int c = 0; c < HD; ++c) {
            float r = hasRes ? Res[ob + c] : b2f(Xin[ob + c]);
            Xout[ob + c] = f2b(fmaxf(out[c] + r, 0.f));
        }
    }
}

// ---------------------------------------------------------------------------
// Final: out[b] = x@Wx + y@Wy + bx + by + sum_j xy[b,j]*y[b,j]
// ---------------------------------------------------------------------------
__global__ __launch_bounds__(256) void final_kernel(
        const ushort_t* __restrict__ X1, const ushort_t* __restrict__ X2,
        const float* __restrict__ xy,
        const float* __restrict__ Wx, const float* __restrict__ Wy,
        const float* __restrict__ bx, const float* __restrict__ by,
        float* __restrict__ out) {
    __shared__ float r1[256], r2[256];
    int b = blockIdx.x, t = threadIdx.x;
    float a1 = 0.f, a2 = 0.f;
    for (int j = t; j < FUSD; j += 256) {
        float xv = b2f(X1[(size_t)b * FUSD + j]);
        float yv = b2f(X2[(size_t)b * FUSD + j]);
        a1 += xv * Wx[j] + yv * Wy[j];
        a2 += xy[(size_t)b * FUSD + j] * yv;
    }
    r1[t] = a1; r2[t] = a2;
    __syncthreads();
    for (int o = 128; o > 0; o >>= 1) {
        if (t < o) { r1[t] += r1[t + o]; r2[t] += r2[t + o]; }
        __syncthreads();
    }
    if (t == 0) out[b] = r1[0] + r2[0] + bx[0] + by[0];
}

// ---------------------------------------------------------------------------
// Host orchestration
// ---------------------------------------------------------------------------
extern "C" void kernel_launch(void* const* d_in, const int* in_sizes, int n_in,
                              void* d_out, int out_size, void* d_ws, size_t ws_size,
                              hipStream_t stream) {
    (void)in_sizes; (void)n_in; (void)out_size; (void)ws_size;

    // -------- inputs (setup_inputs dict insertion order, recursively) --------
    const int*   sparse_ids = (const int*)d_in[0];
    const float* densef     = (const float*)d_in[1];
    const float* emb        = (const float*)d_in[2];
    const float* dW         = (const float*)d_in[3];
    const float* db         = (const float*)d_in[4];
    const float* fs1_ctx    = (const float*)d_in[5];
    const float* fs2_ctx    = (const float*)d_in[6];
    const float* fs1_W1     = (const float*)d_in[7];
    const float* fs1_b1     = (const float*)d_in[8];
    const float* fs1_W2     = (const float*)d_in[9];
    const float* fs1_b2     = (const float*)d_in[10];
    const float* fs2_W1     = (const float*)d_in[11];
    const float* fs2_b1     = (const float*)d_in[12];
    const float* fs2_W2     = (const float*)d_in[13];
    const float* fs2_b2     = (const float*)d_in[14];

    struct LayerP { const float *Wq, *Wk, *Wv, *gw, *gb, *Wres; };
    LayerP msr[2][3];
    int idx = 15;
    for (int br = 0; br < 2; ++br)
        for (int l = 0; l < 3; ++l) {
            msr[br][l].Wq   = (const float*)d_in[idx++];
            msr[br][l].Wk   = (const float*)d_in[idx++];
            msr[br][l].Wv   = (const float*)d_in[idx++];
            msr[br][l].gw   = (const float*)d_in[idx++];
            msr[br][l].gb   = (const float*)d_in[idx++];
            msr[br][l].Wres = (l == 0) ? (const float*)d_in[idx++] : nullptr;
        }
    const float* fus_Wx  = (const float*)d_in[47];
    const float* fus_bx  = (const float*)d_in[48];
    const float* fus_Wy  = (const float*)d_in[49];
    const float* fus_by  = (const float*)d_in[50];
    const float* fus_Wxy = (const float*)d_in[51];
    float* out = (float*)d_out;

    // -------- workspace carve-up --------
    char* ws = (char*)d_ws;
    size_t off = 0;
    auto alloc = [&](size_t bytes) -> void* {
        void* p = ws + off;
        off = (off + bytes + 255) & ~(size_t)255;
        return p;
    };
    float*    g1    = (float*)alloc(128 * sizeof(float));
    float*    g2    = (float*)alloc(128 * sizeof(float));
    float*    stats = (float*)alloc(16 * sizeof(float));
    ushort_t* Wscr  = (ushort_t*)alloc((size_t)256 * 256 * sizeof(ushort_t));
    ushort_t* Xb[2][2];
    for (int br = 0; br < 2; ++br)
        for (int p = 0; p < 2; ++p)
            Xb[br][p] = (ushort_t*)alloc((size_t)BL * D_MSR * sizeof(ushort_t));
    float* Qp  = (float*)alloc((size_t)BL * D_MSR * sizeof(float));
    float* Kp  = (float*)alloc((size_t)BL * D_MSR * sizeof(float));
    float* Vb  = (float*)alloc((size_t)BL * D_MSR * sizeof(float));
    float* Res = (float*)alloc((size_t)BL * D_MSR * sizeof(float));
    ushort_t* WtXY  = (ushort_t*)alloc((size_t)HEADS * HXYPAD * HXY * sizeof(ushort_t));
    float*    xyBuf = (float*)alloc((size_t)B_SZ * FUSD * sizeof(float));

    const float GAMMA[3] = {1.f - 1.f / 32.f, 1.f - 1.f / 128.f, 1.f - 1.f / 512.f};

    // -------- gates (ctx inputs, constant per call) --------
    gate_kernel<<<1, 64, 0, stream>>>(fs1_ctx, fs1_W1, fs1_b1, fs1_W2, fs1_b2, g1);
    gate_kernel<<<1, 64, 0, stream>>>(fs2_ctx, fs2_W1, fs2_b1, fs2_W2, fs2_b2, g2);

    // -------- embedding + dense + gate scaling -> bf16 inputs --------
    embed_kernel<<<BL, 128, 0, stream>>>(sparse_ids, densef, emb, dW, db,
                                         g1, g2, Xb[0][0], Xb[1][0]);

    // projection GEMM helper: N = 256 (no padding needed)
    auto run_gemm = [&](const ushort_t* A, int lda, const float* Wf32,
                        int Kdim, float* C) {
        int nk = 256 * Kdim;
        cvt_tr_kernel<<<(nk + 255) / 256, 256, 0, stream>>>(Wf32, Wscr,
                                                            Kdim, 256, 256);
        dim3 grid(BL / BM, 256 / BN);             // (100, 2)
        gemm_bf16_kernel<<<grid, 256, 0, stream>>>(A, lda, Wscr,
                                                   C, 256, BL, 256, Kdim);
    };

    // -------- MSR stacks --------
    ushort_t* Xfin[2];
    for (int br = 0; br < 2; ++br) {
        ushort_t* Xcur = Xb[br][0];
        ushort_t* Xnxt = Xb[br][1];
        for (int l = 0; l < 3; ++l) {
            int Kdim = (l == 0) ? 128 : 256;
            run_gemm(Xcur, Kdim, msr[br][l].Wq, Kdim, Qp);
            run_gemm(Xcur, Kdim, msr[br][l].Wk, Kdim, Kp);
            run_gemm(Xcur, Kdim, msr[br][l].Wv, Kdim, Vb);
            if (l == 0)
                run_gemm(Xcur, Kdim, msr[br][l].Wres, Kdim, Res);

            xpos_kernel<<<BL, D_MSR, 0, stream>>>(Qp);
            xpos_kernel<<<BL, D_MSR, 0, stream>>>(Kp);

            zero_stats_kernel<<<1, 32, 0, stream>>>(stats);
            gn_stat_kernel<<<BL, D_MSR, 0, stream>>>(Kp, stats);
            gn_apply_kernel<<<BL, D_MSR, 0, stream>>>(Kp, stats,
                                                      msr[br][l].gw, msr[br][l].gb);

            attn_kernel<<<B_SZ * HEADS, 32, 0, stream>>>(
                Qp, Kp, Vb, Res, Xcur, (l == 0) ? 1 : 0, GAMMA[l], Xnxt);

            ushort_t* t = Xcur; Xcur = Xnxt; Xnxt = t;
        }
        Xfin[br] = Xcur;
    }

    // -------- fusion: per-head bilinear (WMMA) + dots --------
    {
        int nk = HXYPAD * HXY;                     // per-head transposed+padded
        for (int h = 0; h < HEADS; ++h)
            cvt_tr_kernel<<<(nk + 255) / 256, 256, 0, stream>>>(
                fus_Wxy + (size_t)h * HXY * HXY,
                WtXY + (size_t)h * HXYPAD * HXY,
                HXY, HXY, HXYPAD);
        for (int h = 0; h < HEADS; ++h) {
            dim3 grid(B_SZ / BM, HXYPAD / BN);     // (4, 7)
            gemm_bf16_kernel<<<grid, 256, 0, stream>>>(
                Xfin[0] + (size_t)h * HXY, FUSD,
                WtXY + (size_t)h * HXYPAD * HXY,
                xyBuf + (size_t)h * HXY, FUSD,
                B_SZ, HXY, HXY);
        }
        final_kernel<<<B_SZ, 256, 0, stream>>>(Xfin[0], Xfin[1], xyBuf,
                                               fus_Wx, fus_Wy, fus_bx, fus_by, out);
    }
}
